// Sc_encoder_2963527434948
// MI455X (gfx1250) — compile-verified, matrix operand order checked by reading, not run on hardware
//
#include <hip/hip_runtime.h>
#include <hip/hip_bf16.h>
#include <math.h>

#define N_NODES 50000
#define DIM     64
#define S1_CNT  20
#define S2_CNT  10
#define GEMM_BLOCKS (N_NODES / 16)   // 3125, exact

typedef __attribute__((ext_vector_type(16))) __bf16 v16bf;
typedef __attribute__((ext_vector_type(8)))  float  v8f;

__device__ __forceinline__ float wave_sum(float v) {
#pragma unroll
    for (int off = 16; off > 0; off >>= 1)
        v += __shfl_xor(v, off, 32);
    return v;
}

// Branch-free tanh: 1 - 2*rcp(exp(2x)+1). Hardware v_rcp_f32 (~1 ulp) instead
// of the IEEE divide ladder; saturates correctly to +/-1 as exp(2x) -> inf / 0.
__device__ __forceinline__ float fast_tanh(float x) {
    return 1.f - 2.f * __builtin_amdgcn_rcpf(__expf(2.f * x) + 1.f);
}

// ---------------------------------------------------------------------------
// Phase 1: intra-path attention. One wave32 per node, lane owns dims [2L,2L+1].
// Online softmax over S neighbors -> single gather pass (256B coalesced rows).
// ---------------------------------------------------------------------------
__global__ __launch_bounds__(256) void intra_att_kernel(
    const float* __restrict__ h,    // (M, 64) neighbor bank
    const float* __restrict__ h0,   // (N, 64) reference embeddings
    const float* __restrict__ att,  // (1, 128) [a_ref | a_nei]
    const int*   __restrict__ nei,  // (N, S)
    float*       __restrict__ e_out,// (N, 64)
    int S)
{
    const int lane = threadIdx.x & 31;
    const int wave = threadIdx.x >> 5;
    const int node = blockIdx.x * 8 + wave;
    if (node >= N_NODES) return;

    const float2 aref = *(const float2*)(att + 2 * lane);
    const float2 anei = *(const float2*)(att + DIM + 2 * lane);
    const float2 h0v  = *(const float2*)(h0 + (size_t)node * DIM + 2 * lane);

    const float r = wave_sum(h0v.x * aref.x + h0v.y * aref.y);

    float  m = -1e30f, l = 0.f;
    float2 acc = make_float2(0.f, 0.f);
    const int* nrow = nei + (size_t)node * S;

    for (int s = 0; s < S; ++s) {
        const int idx = nrow[s];
        const float2 ev = *(const float2*)(h + (size_t)idx * DIM + 2 * lane);
        float p = wave_sum(ev.x * anei.x + ev.y * anei.y) + r;
        p = (p > 0.f) ? p : 0.01f * p;                // leaky_relu(0.01)
        const float nm = fmaxf(m, p);
        const float sc = __expf(m - nm);
        const float w  = __expf(p - nm);
        l     = l * sc + w;
        acc.x = acc.x * sc + w * ev.x;
        acc.y = acc.y * sc + w * ev.y;
        m = nm;
    }
    const float inv = 1.f / l;
    float ex = acc.x * inv, ey = acc.y * inv;
    ex = (ex > 0.f) ? ex : (__expf(ex) - 1.f);        // ELU (alpha=1)
    ey = (ey > 0.f) ? ey : (__expf(ey) - 1.f);
    *(float2*)(e_out + (size_t)node * DIM + 2 * lane) = make_float2(ex, ey);
}

// ---------------------------------------------------------------------------
// Phase 2: tanh(E @ fc_w^T + b) column partial-sums via v_wmma_f32_16x16x32_bf16.
// Block = 16 rows of E; wave w owns output cols [16w,16w+16). K=64 = 2 WMMAs.
// A-operand layout per ISA (16-bit A, wave32): lane L -> m=L%16, half=L/16,
// elems 0..7 : K = half*8 + 0..7 ; elems 8..15 : K = 16 + half*8 + 0..7.
// B-operand = fc_w^T, column j = fc_w row j -> identical addressing pattern.
// C/D layout: lane L -> col = L%16, VGPR r -> row = (L<16 ? r : 8+r).
// ---------------------------------------------------------------------------
__global__ __launch_bounds__(128) void tanh_fc_colsum_kernel(
    const float* __restrict__ E,        // (N, 64)
    const float* __restrict__ W,        // fc_w (64, 64) row-major
    const float* __restrict__ bias_v,   // fc_b (64)
    float*       __restrict__ partials) // (GEMM_BLOCKS, 64)
{
    const int lane = threadIdx.x & 31;
    const int wave = threadIdx.x >> 5;        // 0..3
    const int half = lane >> 4;
    const int sub  = lane & 15;
    const int row0 = blockIdx.x * 16;

    const float* arow = E + (size_t)(row0 + sub) * DIM;       // A row m
    const float* brow = W + (size_t)(wave * 16 + sub) * DIM;  // B col j = W row j

    v8f c = {};
#pragma unroll
    for (int k0 = 0; k0 < DIM; k0 += 32) {
        v16bf a, b;
#pragma unroll
        for (int e = 0; e < 16; ++e) {
            const int k = k0 + ((e < 8) ? 0 : 16) + half * 8 + (e & 7);
            a[e] = (__bf16)arow[k];
            b[e] = (__bf16)brow[k];
        }
        c = __builtin_amdgcn_wmma_f32_16x16x32_bf16(
                /*neg_a=*/false, a, /*neg_b=*/false, b,
                /*c_mod=*/(short)0, c, /*reuse_a=*/false, /*reuse_b=*/false);
    }

    const int   col = wave * 16 + sub;
    const float bv  = bias_v[col];
    float psum = 0.f;
#pragma unroll
    for (int r = 0; r < 8; ++r)
        psum += fast_tanh(c[r] + bv);
    psum += __shfl_xor(psum, 16, 32);   // combine rows 0-7 with rows 8-15
    if (lane < 16)
        partials[(size_t)blockIdx.x * DIM + col] = psum;  // waves own disjoint cols
}

// ---------------------------------------------------------------------------
// Phase 3: deterministic reduction of partials -> sps, beta = softmax(sps@att_i)
// ---------------------------------------------------------------------------
__global__ __launch_bounds__(128) void beta_kernel(
    const float* __restrict__ part1,
    const float* __restrict__ part2,
    const float* __restrict__ att_inter,  // (64)
    float*       __restrict__ beta)       // (2)
{
    __shared__ float s1s[DIM], s2s[DIM];
    const int t = threadIdx.x;
    const int j = t & 63;
    const float* P = (t < 64) ? part1 : part2;
    float s = 0.f;
    for (int b = 0; b < GEMM_BLOCKS; ++b)
        s += P[(size_t)b * DIM + j];
    s = (s / (float)N_NODES) * att_inter[j];
    if (t < 64) s1s[j] = s; else s2s[j] = s;
    __syncthreads();
    if (t == 0) {
        float a = 0.f, bsum = 0.f;
        for (int k = 0; k < DIM; ++k) { a += s1s[k]; bsum += s2s[k]; }
        const float mx = fmaxf(a, bsum);
        const float e0 = __expf(a - mx), e1 = __expf(bsum - mx);
        const float inv = 1.f / (e0 + e1);
        beta[0] = e0 * inv;
        beta[1] = e1 * inv;
    }
}

// ---------------------------------------------------------------------------
// Phase 4: z = e1*beta0 + e2*beta1 (in place over d_out)
// ---------------------------------------------------------------------------
__global__ __launch_bounds__(256) void combine_kernel(
    float*       __restrict__ out,   // e1 in, z out
    const float* __restrict__ e2,
    const float* __restrict__ beta,
    size_t total)
{
    const float b0 = beta[0], b1 = beta[1];
    for (size_t i = (size_t)blockIdx.x * blockDim.x + threadIdx.x; i < total;
         i += (size_t)gridDim.x * blockDim.x)
        out[i] = out[i] * b0 + e2[i] * b1;
}

// ---------------------------------------------------------------------------
extern "C" void kernel_launch(void* const* d_in, const int* in_sizes, int n_in,
                              void* d_out, int out_size, void* d_ws, size_t ws_size,
                              hipStream_t stream)
{
    (void)in_sizes; (void)n_in; (void)out_size; (void)ws_size;

    const float* h0        = (const float*)d_in[0];
    const float* h1        = (const float*)d_in[1];
    const float* h2        = (const float*)d_in[2];
    const float* att1      = (const float*)d_in[3];
    const float* att2      = (const float*)d_in[4];
    const float* fc_w      = (const float*)d_in[5];
    const float* fc_b      = (const float*)d_in[6];
    const float* att_inter = (const float*)d_in[7];
    const int*   nei1      = (const int*)d_in[8];
    const int*   nei2      = (const int*)d_in[9];

    float* e1 = (float*)d_out;                        // (N,64) lives in output
    float* ws = (float*)d_ws;
    float* e2    = ws;                                // N*64 floats
    float* part1 = e2 + (size_t)N_NODES * DIM;        // 3125*64
    float* part2 = part1 + (size_t)GEMM_BLOCKS * DIM; // 3125*64
    float* beta  = part2 + (size_t)GEMM_BLOCKS * DIM; // 2

    const int attn_blocks = N_NODES / 8;              // 6250, exact

    intra_att_kernel<<<attn_blocks, 256, 0, stream>>>(h1, h0, att1, nei1, e1, S1_CNT);
    intra_att_kernel<<<attn_blocks, 256, 0, stream>>>(h2, h0, att2, nei2, e2, S2_CNT);

    tanh_fc_colsum_kernel<<<GEMM_BLOCKS, 128, 0, stream>>>(e1, fc_w, fc_b, part1);
    tanh_fc_colsum_kernel<<<GEMM_BLOCKS, 128, 0, stream>>>(e2, fc_w, fc_b, part2);

    beta_kernel<<<1, 128, 0, stream>>>(part1, part2, att_inter, beta);

    const size_t total = (size_t)N_NODES * DIM;
    combine_kernel<<<12500, 256, 0, stream>>>(e1, e2, beta, total);
}
